// MultiheadAttentionBlock_27788438405386
// MI455X (gfx1250) — matrix operand order repaired and checked by reading, not compiled, not run on hardware
//
#include <hip/hip_runtime.h>
#include <hip/hip_bf16.h>
#include <stdint.h>

// MI455X / gfx1250: wave32, WMMA 16x16x32 bf16->f32 accumulate, TDM staging.
// MHA block: B=4, S=2048, D=1024, H=16, DK=64.

#define Bb 4
#define Ss 2048
#define Dd 1024
#define Hh 16
#define DK 64
#define Mrows (Bb * Ss)   // 8192

typedef __attribute__((ext_vector_type(16))) __bf16          v16bf;
typedef __attribute__((ext_vector_type(16))) unsigned short  v16u;
typedef __attribute__((ext_vector_type(8)))  float           v8f;
typedef __attribute__((ext_vector_type(4)))  unsigned int    v4u;
typedef __attribute__((ext_vector_type(8)))  int             v8i;
typedef __attribute__((ext_vector_type(4)))  int             v4i;

#if __has_builtin(__builtin_amdgcn_tensor_load_to_lds) && \
    __has_builtin(__builtin_amdgcn_s_wait_tensorcnt)
#define HAVE_TDM 1
#else
#define HAVE_TDM 0
#endif

// Native bf16 conversion (hardware cvt, RNE) instead of manual bit twiddling.
__device__ __forceinline__ unsigned short f32_to_bf16(float f) {
    __bf16 h = (__bf16)f;
    return __builtin_bit_cast(unsigned short, h);
}
__device__ __forceinline__ unsigned int pack_bf16x2(float lo, float hi) {
    return (unsigned int)f32_to_bf16(lo) | ((unsigned int)f32_to_bf16(hi) << 16);
}

// A-operand fragment (16xK, 16-bit): lane = M row (via row base pointer),
// lane-half selects K subset. VGPR r<4 -> K=2r(+8*hi); r>=4 -> K=2r+8(+8*hi).
__device__ __forceinline__ v16bf frag_a(const unsigned short* p, int hi) {
    v16u t;
#pragma unroll
    for (int r = 0; r < 8; ++r) {
        int k = (r < 4) ? (2 * r + hi * 8) : (2 * r + 8 + hi * 8);
        unsigned int u = *(const unsigned int*)(p + k);
        t[2 * r]     = (unsigned short)(u & 0xffffu);
        t[2 * r + 1] = (unsigned short)(u >> 16);
    }
    return __builtin_bit_cast(v16bf, t);
}
// B-operand fragment (Kx16, 16-bit): lane = N col (via col base pointer),
// VGPR r<4 -> K=2r(+8*hi); r>=4 -> K=2r+8(+8*hi).
__device__ __forceinline__ v16bf frag_b(const unsigned short* p, int hi) {
    v16u t;
#pragma unroll
    for (int r = 0; r < 8; ++r) {
        unsigned int u = *(const unsigned int*)(p + ((r < 4) ? (2 * r + hi * 8) : (2 * r + 8 + hi * 8)));
        t[2 * r]     = (unsigned short)(u & 0xffffu);
        t[2 * r + 1] = (unsigned short)(u >> 16);
    }
    return __builtin_bit_cast(v16bf, t);
}

__device__ __forceinline__ v8f wmma_bf16(v16bf a, v16bf b, v8f c) {
    // (neg_a, A, neg_b, B, c_mod, C, reuse_a, reuse_b)
    return __builtin_amdgcn_wmma_f32_16x16x32_bf16(false, a, false, b, (short)0, c,
                                                   false, false);
}

// ---------------------------------------------------------------------------
// Tensor Data Mover: 2D tile load Global -> LDS (bf16 elements).
// D# per CDNA5 ISA:
//   group0: count=1, lds_addr, global_addr[56:0], type=2
//   group1: data_size=1 (2B), tensor_dim0/1, tile_dim0/1, tensor_dim0_stride
//   groups 2/3 (and extra group on this toolchain) zero (<=2D tensor), cpol 0.
// ---------------------------------------------------------------------------
__device__ __forceinline__ void tdm_load_2d_bf16(unsigned lds_off, const void* gaddr,
                                                 unsigned dim0, unsigned dim1,
                                                 unsigned tile0, unsigned tile1,
                                                 unsigned stride0) {
#if HAVE_TDM
    unsigned long long ga = (unsigned long long)(uintptr_t)gaddr;
    v4u g0;
    g0[0] = 1u;                                                  // count=1
    g0[1] = lds_off;                                             // lds_addr (bytes)
    g0[2] = (unsigned)(ga & 0xffffffffu);                        // global_addr[31:0]
    g0[3] = (unsigned)((ga >> 32) & 0x1ffffffu) | (2u << 30);    // [56:32] | type=2
    v8i g1;
    g1[0] = (int)(1u << 16);                                     // data_size=1 -> 2B
    g1[1] = (int)((dim0 & 0xffffu) << 16);                       // tensor_dim0 lo
    g1[2] = (int)(((dim0 >> 16) & 0xffffu) | ((dim1 & 0xffffu) << 16));
    g1[3] = (int)(((dim1 >> 16) & 0xffffu) | ((tile0 & 0xffffu) << 16));
    g1[4] = (int)(tile1 & 0xffffu);                              // tile_dim1, dim2=0
    g1[5] = (int)stride0;                                        // dim0_stride lo
    g1[6] = 0;
    g1[7] = 0;
    v4i z4 = {0, 0, 0, 0};
    v8i z8 = {0, 0, 0, 0, 0, 0, 0, 0};
    // 6-arg form (clang-23 / therock headers): g0, g1, g2, g3, extra, cpol
    __builtin_amdgcn_tensor_load_to_lds(g0, g1, z4, z4, z8, 0);
#else
    (void)lds_off; (void)gaddr; (void)dim0; (void)dim1;
    (void)tile0; (void)tile1; (void)stride0;
#endif
}
__device__ __forceinline__ void tdm_wait0() {
#if HAVE_TDM
    __builtin_amdgcn_s_wait_tensorcnt(0);
#endif
}
__device__ __forceinline__ unsigned lds_offset_of(const void* p) {
    return (unsigned)(uintptr_t)p;   // flat LDS aperture: addr[31:0] = LDS byte addr
}

// ---------------------------------------------------------------------------
// Projection GEMM: Out[b,h,s,dk](bf16) = A[8192,1024](f32) @ W[1024,1024]^T(f32)
// Block tile 128(M) x 64(N), 8 waves -> each wave 32x32 via 2x2 WMMA accum.
// f32 -> bf16 conversion folded into the LDS staging (native cvt).
// ---------------------------------------------------------------------------
__global__ __launch_bounds__(256) void proj_gemm_kernel(
    const float* __restrict__ A, const float* __restrict__ W,
    unsigned short* __restrict__ Out) {
    __shared__ unsigned short ldsA[128 * 32];  // [m][k] bf16
    __shared__ unsigned short ldsB[64 * 32];   // [n][k] bf16

    const int tid = threadIdx.x;
    const int lane = tid & 31, wv = tid >> 5;
    const int lane15 = lane & 15, hi = lane >> 4;
    const int wm = (wv & 3) * 32, wn = (wv >> 2) * 32;
    const int blockN = blockIdx.x * 64, blockM = blockIdx.y * 128;

    v8f acc[2][2] = {{{}, {}}, {{}, {}}};

    for (int k0 = 0; k0 < Dd; k0 += 32) {
        {   // stage A tile 128x32 f32 -> bf16 (16 elems/thread)
            int row = tid >> 1, colb = (tid & 1) * 16;
            const float* src = A + (size_t)(blockM + row) * Dd + k0 + colb;
            float4 f0 = ((const float4*)src)[0];
            float4 f1 = ((const float4*)src)[1];
            float4 f2 = ((const float4*)src)[2];
            float4 f3 = ((const float4*)src)[3];
            uint4 u0, u1;
            u0.x = pack_bf16x2(f0.x, f0.y); u0.y = pack_bf16x2(f0.z, f0.w);
            u0.z = pack_bf16x2(f1.x, f1.y); u0.w = pack_bf16x2(f1.z, f1.w);
            u1.x = pack_bf16x2(f2.x, f2.y); u1.y = pack_bf16x2(f2.z, f2.w);
            u1.z = pack_bf16x2(f3.x, f3.y); u1.w = pack_bf16x2(f3.z, f3.w);
            *(uint4*)&ldsA[row * 32 + colb]     = u0;
            *(uint4*)&ldsA[row * 32 + colb + 8] = u1;
        }
        {   // stage W tile 64x32 f32 -> bf16 (8 elems/thread)
            int row = tid >> 2, colb = (tid & 3) * 8;
            const float* src = W + (size_t)(blockN + row) * Dd + k0 + colb;
            float4 f0 = ((const float4*)src)[0];
            float4 f1 = ((const float4*)src)[1];
            uint4 u;
            u.x = pack_bf16x2(f0.x, f0.y); u.y = pack_bf16x2(f0.z, f0.w);
            u.z = pack_bf16x2(f1.x, f1.y); u.w = pack_bf16x2(f1.z, f1.w);
            *(uint4*)&ldsB[row * 32 + colb] = u;
        }
        __syncthreads();

        v16bf bfr[2];
#pragma unroll
        for (int j = 0; j < 2; ++j)
            bfr[j] = frag_b(&ldsB[(wn + j * 16 + lane15) * 32], hi);
#pragma unroll
        for (int i = 0; i < 2; ++i) {
            v16bf a = frag_a(&ldsA[(wm + i * 16 + lane15) * 32], hi);
#pragma unroll
            for (int j = 0; j < 2; ++j) acc[i][j] = wmma_bf16(a, bfr[j], acc[i][j]);
        }
        __syncthreads();
    }

    // write-out in head-split layout [B,H,S,DK] as bf16
#pragma unroll
    for (int i = 0; i < 2; ++i)
#pragma unroll
        for (int j = 0; j < 2; ++j)
#pragma unroll
            for (int r = 0; r < 8; ++r) {
                int row = blockM + wm + i * 16 + r + hi * 8;   // global token row
                int col = blockN + wn + j * 16 + lane15;       // feature index
                int b = row >> 11, q = row & (Ss - 1);
                int h = col >> 6, dk = col & (DK - 1);
                Out[(((size_t)(b * Hh + h)) * Ss + q) * DK + dk] =
                    f32_to_bf16(acc[i][j][r]);
            }
}

// ---------------------------------------------------------------------------
// Flash attention: one block per (b, h, 128-query tile); 8 waves x 16 q-rows.
// K tile staged via TDM (contiguous 4KB), V transposed manually (TDM cannot
// transpose); online softmax; all matmuls WMMA.
// Q/K/V in [B,H,S,64] bf16; X out in [B,S,H*64] bf16.
// ---------------------------------------------------------------------------
__global__ __launch_bounds__(256) void flash_attn_kernel(
    const unsigned short* __restrict__ Qw, const unsigned short* __restrict__ Kw,
    const unsigned short* __restrict__ Vw, unsigned short* __restrict__ Xw) {
    __shared__ unsigned short ldsK[32 * 64];       // [key][dk]
    __shared__ unsigned short ldsVt[64 * 32];      // [dk][key] (transposed)
    __shared__ unsigned short ldsP[8 * 16 * 32];   // per-wave probs [16][32]

    const int tid = threadIdx.x;
    const int lane = tid & 31, wv = tid >> 5;
    const int lane15 = lane & 15, hi = lane >> 4;

    const int qTiles = Ss / 128;                    // 16
    const int bh = blockIdx.x / qTiles;             // 0..63
    const int qt = blockIdx.x % qTiles;
    const int b = bh / Hh, h = bh % Hh;

    const size_t bhBase = (size_t)bh * Ss * DK;
    const int qbase = qt * 128 + wv * 16;           // this wave's first q row
    const unsigned ldsK_off = lds_offset_of(&ldsK[0]);

    // Q fragments (16 rows x 64 dk) loaded once, straight from global
    const unsigned short* qrow = Qw + bhBase + (size_t)(qbase + lane15) * DK;
    v16bf aq0 = frag_a(qrow, hi);
    v16bf aq1 = frag_a(qrow + 32, hi);

    v8f o[4] = {{}, {}, {}, {}};
    float m_run[8], l_run[8];
#pragma unroll
    for (int r = 0; r < 8; ++r) { m_run[r] = -3.0e38f; l_run[r] = 0.0f; }

    unsigned short* pls = &ldsP[wv * 16 * 32];
    const float sm_scale = 0.125f;                  // 1/sqrt(64)

    for (int kt = 0; kt < Ss / 32; ++kt) {
#if HAVE_TDM
        if (wv == 0) {  // K tile: 32 keys x 64 dk, contiguous -> 1x2048 TDM tile
            tdm_load_2d_bf16(ldsK_off, Kw + bhBase + (size_t)kt * 32 * DK,
                             /*dim0=*/2048, /*dim1=*/1, /*tile0=*/2048, /*tile1=*/1,
                             /*stride0=*/2048);
        }
#else
        {   // stage K tile 32x64 (uint4 = 8 bf16 per thread)
            int row = tid >> 3, colb = (tid & 7) * 8;
            const unsigned short* src = Kw + bhBase + (size_t)(kt * 32 + row) * DK + colb;
            *(uint4*)&ldsK[row * 64 + colb] = *(const uint4*)src;
        }
#endif
        {   // stage V tile transposed: [dk][key]
            int key = tid >> 3, colb = (tid & 7) * 8;
            const unsigned short* src = Vw + bhBase + (size_t)(kt * 32 + key) * DK + colb;
            uint4 u = *(const uint4*)src;
            unsigned int w4[4] = {u.x, u.y, u.z, u.w};
#pragma unroll
            for (int c = 0; c < 8; ++c) {
                unsigned short e = (unsigned short)((w4[c >> 1] >> ((c & 1) * 16)) & 0xffffu);
                ldsVt[(colb + c) * 32 + key] = e;
            }
        }
        if (kt + 1 < Ss / 32) {  // prefetch next V tile (global_prefetch_b8)
            int key = tid >> 3;
            __builtin_prefetch(Vw + bhBase + (size_t)((kt + 1) * 32 + key) * DK, 0, 1);
        }
#if HAVE_TDM
        if (wv == 0) tdm_wait0();  // K tile landed before barrier release
#endif
        __syncthreads();

        // scores S[16x32] = Q(16x64) @ K^T : 4 WMMAs
        v8f sc0 = {}, sc1 = {};
#pragma unroll
        for (int ks = 0; ks < 2; ++ks) {
            v16bf a = ks ? aq1 : aq0;
            v16bf b0 = frag_b(&ldsK[lane15 * 64 + ks * 32], hi);            // keys 0..15
            v16bf b1 = frag_b(&ldsK[(16 + lane15) * 64 + ks * 32], hi);     // keys 16..31
            sc0 = wmma_bf16(a, b0, sc0);
            sc1 = wmma_bf16(a, b1, sc1);
        }

        // online softmax per row (row = r + hi*8; 16 lanes per row)
        float corr[8];
#pragma unroll
        for (int r = 0; r < 8; ++r) {
            float s0 = sc0[r] * sm_scale, s1 = sc1[r] * sm_scale;
            float mx = fmaxf(s0, s1);
#pragma unroll
            for (int msk = 1; msk < 16; msk <<= 1) mx = fmaxf(mx, __shfl_xor(mx, msk, 32));
            float m_new = fmaxf(m_run[r], mx);
            corr[r] = __expf(m_run[r] - m_new);
            m_run[r] = m_new;
            float p0 = __expf(s0 - m_new), p1 = __expf(s1 - m_new);
            float rs = p0 + p1;
#pragma unroll
            for (int msk = 1; msk < 16; msk <<= 1) rs += __shfl_xor(rs, msk, 32);
            l_run[r] = l_run[r] * corr[r] + rs;
            int prow = r + hi * 8;
            pls[prow * 32 + lane15]      = f32_to_bf16(p0);
            pls[prow * 32 + 16 + lane15] = f32_to_bf16(p1);
        }
#pragma unroll
        for (int n = 0; n < 4; ++n)
#pragma unroll
            for (int r = 0; r < 8; ++r) o[n][r] *= corr[r];

        // O += P(16x32) @ V(32x64): 4 WMMAs (per-wave LDS round-trip relays P
        // from C/D layout to A layout; same-wave DS ops stay ordered)
        v16bf ap = frag_a(&pls[lane15 * 32], hi);
#pragma unroll
        for (int n = 0; n < 4; ++n) {
            v16bf bv = frag_b(&ldsVt[(n * 16 + lane15) * 32], hi);
            o[n] = wmma_bf16(ap, bv, o[n]);
        }
        __syncthreads();
    }

    // normalize and write X in [B,S,H*DK] bf16
#pragma unroll
    for (int r = 0; r < 8; ++r) {
        float inv = 1.0f / l_run[r];
        int q = qbase + r + hi * 8;
        size_t rowbase = ((size_t)(b * Ss + q)) * Dd + h * DK;
#pragma unroll
        for (int n = 0; n < 4; ++n)
            Xw[rowbase + n * 16 + lane15] = f32_to_bf16(o[n][r] * inv);
    }
}

// ---------------------------------------------------------------------------
// Output GEMM: out[8192,1024](f32) = X[8192,1024](bf16) @ W_O[1024,1024]^T(f32)
// X tile (128x32 bf16, stride 1024) staged via TDM; W_O converted manually.
// ---------------------------------------------------------------------------
__global__ __launch_bounds__(256) void out_gemm_kernel(
    const unsigned short* __restrict__ X, const float* __restrict__ W,
    float* __restrict__ Out) {
    __shared__ unsigned short ldsA[128 * 32];
    __shared__ unsigned short ldsB[64 * 32];

    const int tid = threadIdx.x;
    const int lane = tid & 31, wv = tid >> 5;
    const int lane15 = lane & 15, hi = lane >> 4;
    const int wm = (wv & 3) * 32, wn = (wv >> 2) * 32;
    const int blockN = blockIdx.x * 64, blockM = blockIdx.y * 128;
    const unsigned ldsA_off = lds_offset_of(&ldsA[0]);

    v8f acc[2][2] = {{{}, {}}, {{}, {}}};

    for (int k0 = 0; k0 < Dd; k0 += 32) {
#if HAVE_TDM
        if (wv == 0) {  // X tile: 128 rows x 32 cols bf16, row stride 1024
            tdm_load_2d_bf16(ldsA_off, X + (size_t)blockM * Dd + k0,
                             /*dim0=*/32, /*dim1=*/128, /*tile0=*/32, /*tile1=*/128,
                             /*stride0=*/Dd);
        }
#else
        {   // stage X tile (already bf16): 2x uint4 per thread
            int row = tid >> 1, colb = (tid & 1) * 16;
            const unsigned short* src = X + (size_t)(blockM + row) * Dd + k0 + colb;
            uint4 u0 = ((const uint4*)src)[0];
            uint4 u1 = ((const uint4*)src)[1];
            *(uint4*)&ldsA[row * 32 + colb]     = u0;
            *(uint4*)&ldsA[row * 32 + colb + 8] = u1;
        }
#endif
        {   // stage W_O tile with f32 -> bf16 conversion
            int row = tid >> 2, colb = (tid & 3) * 8;
            const float* src = W + (size_t)(blockN + row) * Dd + k0 + colb;
            float4 f0 = ((const float4*)src)[0];
            float4 f1 = ((const float4*)src)[1];
            uint4 u;
            u.x = pack_bf16x2(f0.x, f0.y); u.y = pack_bf16x2(f0.z, f0.w);
            u.z = pack_bf16x2(f1.x, f1.y); u.w = pack_bf16x2(f1.z, f1.w);
            *(uint4*)&ldsB[row * 32 + colb] = u;
        }
#if HAVE_TDM
        if (wv == 0) tdm_wait0();
#endif
        __syncthreads();

        v16bf bfr[2];
#pragma unroll
        for (int j = 0; j < 2; ++j)
            bfr[j] = frag_b(&ldsB[(wn + j * 16 + lane15) * 32], hi);
#pragma unroll
        for (int i = 0; i < 2; ++i) {
            v16bf a = frag_a(&ldsA[(wm + i * 16 + lane15) * 32], hi);
#pragma unroll
            for (int j = 0; j < 2; ++j) acc[i][j] = wmma_bf16(a, bfr[j], acc[i][j]);
        }
        __syncthreads();
    }

#pragma unroll
    for (int i = 0; i < 2; ++i)
#pragma unroll
        for (int j = 0; j < 2; ++j)
#pragma unroll
            for (int r = 0; r < 8; ++r) {
                int row = blockM + wm + i * 16 + r + hi * 8;
                int col = blockN + wn + j * 16 + lane15;
                Out[(size_t)row * Dd + col] = acc[i][j][r];
            }
}

// ---------------------------------------------------------------------------
extern "C" void kernel_launch(void* const* d_in, const int* in_sizes, int n_in,
                              void* d_out, int out_size, void* d_ws, size_t ws_size,
                              hipStream_t stream) {
    (void)in_sizes; (void)n_in; (void)out_size; (void)ws_size;
    const float* q   = (const float*)d_in[0];
    const float* k   = (const float*)d_in[1];
    const float* v   = (const float*)d_in[2];
    // d_in[3] = mask (unused by reference)
    const float* W_Q = (const float*)d_in[4];
    const float* W_K = (const float*)d_in[5];
    const float* W_V = (const float*)d_in[6];
    const float* W_O = (const float*)d_in[7];
    float* out = (float*)d_out;

    // workspace layout (bf16): Qw | Kw | Vw | Xw, 16 MB each (64 MB total)
    const size_t elems = (size_t)Mrows * Dd;
    unsigned short* Qw = (unsigned short*)d_ws;
    unsigned short* Kw = Qw + elems;
    unsigned short* Vw = Kw + elems;
    unsigned short* Xw = Vw + elems;

    dim3 blk(256);
    dim3 gGemm(Dd / 64, Mrows / 128);          // (16, 64)
    dim3 gAttn(Bb * Hh * (Ss / 128));          // 1024 blocks

    proj_gemm_kernel<<<gGemm, blk, 0, stream>>>(q, W_Q, Qw);
    proj_gemm_kernel<<<gGemm, blk, 0, stream>>>(k, W_K, Kw);
    proj_gemm_kernel<<<gGemm, blk, 0, stream>>>(v, W_V, Vw);
    flash_attn_kernel<<<gAttn, blk, 0, stream>>>(Qw, Kw, Vw, Xw);
    out_gemm_kernel<<<gGemm, blk, 0, stream>>>(Xw, W_O, out);
}
